// GroupedQueryAttention_5634997093192
// MI455X (gfx1250) — matrix operand order repaired and checked by reading, not compiled, not run on hardware
//
#include <hip/hip_runtime.h>

typedef __bf16 bf16;
typedef __attribute__((ext_vector_type(16))) __bf16 v16bf;
typedef __attribute__((ext_vector_type(8)))  __bf16 bf16x8;
typedef __attribute__((ext_vector_type(8)))  float  v8f;

#define NUM_HEADS 16
#define HEAD_DIM  128
#define DM        2048
#define KVD       512
#define BATCH     2
#define SEQ       2048
#define MTOT      (BATCH * SEQ)

__device__ __forceinline__ v16bf make_frag(const bf16* lo, const bf16* hi) {
  bf16x8 a = *(const bf16x8*)lo;
  bf16x8 b = *(const bf16x8*)hi;
  v16bf r;
#pragma unroll
  for (int i = 0; i < 8; ++i) { r[i] = a[i]; r[i + 8] = b[i]; }
  return r;
}

// gfx1250 async global->LDS copy (16B per lane), tracked by ASYNCcnt.
__device__ __forceinline__ void async_copy_b128(unsigned lds_off, const void* gptr) {
  asm volatile("global_load_async_to_lds_b128 %0, %1, off"
               :: "v"(lds_off), "v"(gptr) : "memory");
}
__device__ __forceinline__ void wait_async0() {
  asm volatile("s_wait_asynccnt 0" ::: "memory");
}
__device__ __forceinline__ unsigned lds_off32(const void* p) {
  return (unsigned)(unsigned long long)p;  // low 32 bits of generic addr = LDS offset
}

// C[M,N] = A[M,K] @ W[N,K]^T + bias[N]
// A f32 or bf16 (ABF16); C stored bf16 (OUTBF16) or f32.
// Block tile 128x128, 8 waves as 4(M)x2(N); wave tile 32x64.
// Double-buffered LDS: one barrier per 32-wide K slab.
template <bool ABF16, bool OUTBF16>
__global__ __launch_bounds__(256) void gemm_wmma(
    const void* __restrict__ Av, const float* __restrict__ W,
    const float* __restrict__ bias, void* __restrict__ Cv,
    int M, int N, int K) {
  constexpr int LDT = 40;  // padded row stride (halves): conflict-free b128 reads
  __shared__ bf16 lsA[2][128][LDT];
  __shared__ bf16 lsB[2][128][LDT];

  const int tid  = threadIdx.x;
  const int wave = tid >> 5;
  const int lane = tid & 31;
  const int lh   = lane & 15;
  const int hsel = lane >> 4;
  const int m0   = (wave >> 1) * 32;  // wave M offset in block tile
  const int n0   = (wave & 1) * 64;   // wave N offset in block tile
  const int mBase = blockIdx.x * 128;
  const int nBase = blockIdx.y * 128;

  const int ldRow = tid >> 1;        // 0..127
  const int ldCol = (tid & 1) * 16;  // 0 or 16

  v8f acc[8] = {};       // [0..3]: rows m0..m0+15, [4..7]: rows m0+16..m0+31
  float4 ar[4], wr[4];   // register staging for f32 tiles

  auto issue_loads = [&](int buf, int kt) {
    if constexpr (ABF16) {
      // bf16 A: pure copy -> async straight into LDS (ASYNCcnt path)
      const bf16* src = (const bf16*)Av + (size_t)(mBase + ldRow) * K + kt + ldCol;
      async_copy_b128(lds_off32(&lsA[buf][ldRow][ldCol]), src);
      async_copy_b128(lds_off32(&lsA[buf][ldRow][ldCol + 8]), src + 8);
    } else {
      const float4* s4 =
          (const float4*)((const float*)Av + (size_t)(mBase + ldRow) * K + kt + ldCol);
#pragma unroll
      for (int q = 0; q < 4; ++q) ar[q] = s4[q];
    }
    const float4* w4 = (const float4*)(W + (size_t)(nBase + ldRow) * K + kt + ldCol);
#pragma unroll
    for (int q = 0; q < 4; ++q) wr[q] = w4[q];
  };

  auto store_tiles = [&](int buf) {
    if constexpr (!ABF16) {
#pragma unroll
      for (int q = 0; q < 4; ++q) {
        lsA[buf][ldRow][ldCol + 4 * q + 0] = (bf16)ar[q].x;
        lsA[buf][ldRow][ldCol + 4 * q + 1] = (bf16)ar[q].y;
        lsA[buf][ldRow][ldCol + 4 * q + 2] = (bf16)ar[q].z;
        lsA[buf][ldRow][ldCol + 4 * q + 3] = (bf16)ar[q].w;
      }
    }
#pragma unroll
    for (int q = 0; q < 4; ++q) {
      lsB[buf][ldRow][ldCol + 4 * q + 0] = (bf16)wr[q].x;
      lsB[buf][ldRow][ldCol + 4 * q + 1] = (bf16)wr[q].y;
      lsB[buf][ldRow][ldCol + 4 * q + 2] = (bf16)wr[q].z;
      lsB[buf][ldRow][ldCol + 4 * q + 3] = (bf16)wr[q].w;
    }
  };

  auto compute = [&](int buf) {
    const int koff = hsel * 8;  // lanes 0-15: K {0..7,16..23}; 16-31: {8..15,24..31}
    v16bf af0 = make_frag(&lsA[buf][m0 + lh][koff], &lsA[buf][m0 + lh][koff + 16]);
    v16bf af1 = make_frag(&lsA[buf][m0 + 16 + lh][koff], &lsA[buf][m0 + 16 + lh][koff + 16]);
#pragma unroll
    for (int j = 0; j < 4; ++j) {
      const int brow = n0 + j * 16 + lh;  // col n; lanes 0-15 K {0..7,16..23}, 16-31 K {8..15,24..31}
      v16bf bfr = make_frag(&lsB[buf][brow][hsel * 8], &lsB[buf][brow][hsel * 8 + 16]);
      acc[j]     = __builtin_amdgcn_wmma_f32_16x16x32_bf16(false, af0, false, bfr,
                                                           (short)0, acc[j], false, false);
      acc[4 + j] = __builtin_amdgcn_wmma_f32_16x16x32_bf16(false, af1, false, bfr,
                                                           (short)0, acc[4 + j], false, false);
    }
  };

  // prologue: fill buffer 0
  issue_loads(0, 0);
  store_tiles(0);
  if constexpr (ABF16) wait_async0();
  __syncthreads();

  int buf = 0;
  for (int kt = 32; kt < K; kt += 32) {
    issue_loads(buf ^ 1, kt);  // fetch next slab (async goes straight to LDS)
    compute(buf);              // WMMA on current slab
    store_tiles(buf ^ 1);      // converted f32 tiles into other buffer
    if constexpr (ABF16) wait_async0();
    __syncthreads();           // single barrier per K slab
    buf ^= 1;
  }
  compute(buf);

  // epilogue: bias + store (C layout: elem i -> row i+8*hsel, col lh)
#pragma unroll
  for (int j = 0; j < 4; ++j) {
    const int col  = nBase + n0 + j * 16 + lh;
    const float bv = bias[col];
#pragma unroll
    for (int i = 0; i < 8; ++i) {
      const int row0 = mBase + m0 + i + 8 * hsel;
      const float v0 = acc[j][i] + bv;
      const float v1 = acc[4 + j][i] + bv;
      if constexpr (OUTBF16) {
        ((bf16*)Cv)[(size_t)row0 * N + col]        = (bf16)v0;
        ((bf16*)Cv)[(size_t)(row0 + 16) * N + col] = (bf16)v1;
      } else {
        ((float*)Cv)[(size_t)row0 * N + col]        = v0;
        ((float*)Cv)[(size_t)(row0 + 16) * N + col] = v1;
      }
    }
  }
}

// Flash-attention, causal, GQA (4 query heads share each KV head).
// Q,O: [B,S,2048] bf16; K,V: [B,S,512] bf16. One block = 128 queries of one head.
__global__ __launch_bounds__(256) void gqa_attn(
    const bf16* __restrict__ Q, const bf16* __restrict__ Kb,
    const bf16* __restrict__ Vb, bf16* __restrict__ O) {
  constexpr int LKT = HEAD_DIM + 8;  // 136 halves = 272B rows
  constexpr int LVT = 40;
  __shared__ bf16 lsK[32][LKT];         // K chunk, [t][h]
  __shared__ bf16 lsVT[HEAD_DIM][LVT];  // V chunk transposed, [h][t]
  __shared__ bf16 lsP[8][16][LVT];      // per-wave probs scratch (C-layout -> A-frag)

  const int tid  = threadIdx.x;
  const int wave = tid >> 5;
  const int lane = tid & 31;
  const int lh   = lane & 15;
  const int hsel = lane >> 4;

  const int nQblk = SEQ / 128;  // 16
  int bid = blockIdx.x;
  const int qblk = bid % nQblk; bid /= nQblk;
  const int head = bid % NUM_HEADS; bid /= NUM_HEADS;
  const int b    = bid;
  const int kv   = head >> 2;

  const int q0 = qblk * 128 + wave * 16;  // this wave's first query
  const size_t qbase = (size_t)b * SEQ * DM;

  // Q fragments (16 x 128, four 32-wide K slabs over head_dim)
  v16bf qf[4];
  {
    const bf16* qp = Q + qbase + (size_t)(q0 + lh) * DM + head * HEAD_DIM;
#pragma unroll
    for (int ks = 0; ks < 4; ++ks) {
      const bf16* p = qp + ks * 32 + hsel * 8;
      qf[ks] = make_frag(p, p + 16);
    }
  }

  float m_run[8], l_run[8];
  v8f oacc[8] = {};
#pragma unroll
  for (int i = 0; i < 8; ++i) { m_run[i] = -3.0e38f; l_run[i] = 0.0f; }

  const float scale = 0.08838834764831845f;  // 1/sqrt(128)

  const int ldT = tid >> 3;        // 0..31 (key within chunk)
  const int ldH = (tid & 7) * 16;  // 0..112 (head-dim group)

  const int tEnd = qblk * 128 + 128;
  for (int t0 = 0; t0 < tEnd; t0 += 32) {
    // cooperative stage: K chunk via async global->LDS, V chunk transposed via VALU
    {
      const size_t rb = ((size_t)b * SEQ + (t0 + ldT)) * KVD + kv * HEAD_DIM + ldH;
      async_copy_b128(lds_off32(&lsK[ldT][ldH]), Kb + rb);
      async_copy_b128(lds_off32(&lsK[ldT][ldH + 8]), Kb + rb + 8);
      bf16x8 v0 = *(const bf16x8*)(Vb + rb);
      bf16x8 v1 = *(const bf16x8*)(Vb + rb + 8);
#pragma unroll
      for (int e = 0; e < 8; ++e) {
        lsVT[ldH + e][ldT]     = v0[e];
        lsVT[ldH + 8 + e][ldT] = v1[e];
      }
    }
    wait_async0();
    __syncthreads();

    if (t0 <= q0 + 15) {  // causal: chunk intersects this wave's rows
      // scores S[16 x 32] = Q @ K^T
      v8f s0 = {}, s1 = {};
#pragma unroll
      for (int ks = 0; ks < 4; ++ks) {
        v16bf bk0 = make_frag(&lsK[lh][ks * 32 + hsel * 8],
                              &lsK[lh][ks * 32 + hsel * 8 + 16]);
        v16bf bk1 = make_frag(&lsK[16 + lh][ks * 32 + hsel * 8],
                              &lsK[16 + lh][ks * 32 + hsel * 8 + 16]);
        s0 = __builtin_amdgcn_wmma_f32_16x16x32_bf16(false, qf[ks], false, bk0,
                                                     (short)0, s0, false, false);
        s1 = __builtin_amdgcn_wmma_f32_16x16x32_bf16(false, qf[ks], false, bk1,
                                                     (short)0, s1, false, false);
      }
      // online softmax in C-register layout
#pragma unroll
      for (int i = 0; i < 8; ++i) {
        const int row = q0 + i + 8 * hsel;  // global query index
        float e0 = s0[i] * scale;
        float e1 = s1[i] * scale;
        if (t0 + lh > row)      e0 = -3.0e38f;
        if (t0 + 16 + lh > row) e1 = -3.0e38f;
        float mx = fmaxf(e0, e1);
#pragma unroll
        for (int off = 1; off < 16; off <<= 1)
          mx = fmaxf(mx, __shfl_xor(mx, off, 32));
        const float mnew  = fmaxf(m_run[i], mx);
        const float alpha = __expf(m_run[i] - mnew);
        const float p0 = __expf(e0 - mnew);
        const float p1 = __expf(e1 - mnew);
        float ps = p0 + p1;
#pragma unroll
        for (int off = 1; off < 16; off <<= 1)
          ps += __shfl_xor(ps, off, 32);
        l_run[i] = l_run[i] * alpha + ps;
        m_run[i] = mnew;
#pragma unroll
        for (int j = 0; j < 8; ++j) oacc[j][i] *= alpha;
        const int pr = i + 8 * hsel;
        lsP[wave][pr][lh]      = (bf16)p0;
        lsP[wave][pr][16 + lh] = (bf16)p1;
      }
      asm volatile("" ::: "memory");  // same-wave LDS is in-order; block reordering
      // O += P @ V
      v16bf pf = make_frag(&lsP[wave][lh][hsel * 8], &lsP[wave][lh][hsel * 8 + 16]);
#pragma unroll
      for (int j = 0; j < 8; ++j) {
        v16bf bv = make_frag(&lsVT[j * 16 + lh][hsel * 8],
                             &lsVT[j * 16 + lh][hsel * 8 + 16]);
        oacc[j] = __builtin_amdgcn_wmma_f32_16x16x32_bf16(false, pf, false, bv,
                                                          (short)0, oacc[j], false, false);
      }
    }
    __syncthreads();
  }

  // normalize and write O (bf16)
#pragma unroll
  for (int i = 0; i < 8; ++i) {
    const float inv = 1.0f / l_run[i];
    const int s = q0 + i + 8 * hsel;
    bf16* op = O + qbase + (size_t)s * DM + head * HEAD_DIM;
#pragma unroll
    for (int j = 0; j < 8; ++j)
      op[j * 16 + lh] = (bf16)(oacc[j][i] * inv);
  }
}

extern "C" void kernel_launch(void* const* d_in, const int* in_sizes, int n_in,
                              void* d_out, int out_size, void* d_ws, size_t ws_size,
                              hipStream_t stream) {
  (void)in_sizes; (void)n_in; (void)out_size; (void)ws_size;
  const float* x   = (const float*)d_in[0];
  const float* W_q = (const float*)d_in[1];
  const float* b_q = (const float*)d_in[2];
  const float* W_k = (const float*)d_in[3];
  const float* b_k = (const float*)d_in[4];
  const float* W_v = (const float*)d_in[5];
  const float* b_v = (const float*)d_in[6];
  const float* W_o = (const float*)d_in[7];
  const float* b_o = (const float*)d_in[8];
  float* out = (float*)d_out;

  // workspace: Q (16.8MB) + K (4.2MB) + V (4.2MB) + O (16.8MB), all bf16
  bf16* Qb = (bf16*)d_ws;
  bf16* Kb = Qb + (size_t)MTOT * DM;
  bf16* Vb = Kb + (size_t)MTOT * KVD;
  bf16* Ob = Vb + (size_t)MTOT * KVD;

  dim3 blk(256);
  gemm_wmma<false, true><<<dim3(MTOT / 128, DM / 128), blk, 0, stream>>>(
      x, W_q, b_q, Qb, MTOT, DM, DM);
  gemm_wmma<false, true><<<dim3(MTOT / 128, KVD / 128), blk, 0, stream>>>(
      x, W_k, b_k, Kb, MTOT, KVD, DM);
  gemm_wmma<false, true><<<dim3(MTOT / 128, KVD / 128), blk, 0, stream>>>(
      x, W_v, b_v, Vb, MTOT, KVD, DM);
  gqa_attn<<<dim3(BATCH * NUM_HEADS * (SEQ / 128)), blk, 0, stream>>>(Qb, Kb, Vb, Ob);
  gemm_wmma<true, false><<<dim3(MTOT / 128, DM / 128), blk, 0, stream>>>(
      Ob, W_o, b_o, out, MTOT, DM, DM);
}
